// ISIR_61186104099357
// MI455X (gfx1250) — matrix operand order repaired and checked by reading, not compiled, not run on hardware
//
#include <hip/hip_runtime.h>
#include <hip/hip_bf16.h>
#include <math.h>

// ISIR on MI455X (gfx1250):
//   K1: f16 WMMA GEMM (z-mu)@A + rowwise ||y||^2, ||eps||^2 -> proposal log-weights
//       (dual-tile per wave: each LDS B-fragment feeds two v_wmma ops so the
//        WMMA rate is not capped by the 256 B/clk LDS bank bandwidth)
//   K2: per-chain sequential gumbel-argmax scan -> source codes + n_eff
//   K3: materialize samples = q_loc + exp(qls)*eps[src]
// Problem dims hardcoded per setup_inputs(): n=16, K=32, nc=1024, d=256.

typedef __attribute__((ext_vector_type(16))) _Float16 v16h;
typedef __attribute__((ext_vector_type(8)))  float    v8f;

#define NSTEP 16
#define KM1   31
#define NC    1024
#define DDIM  256

// ---------------------------------------------------------------------------
// Kernel 0: convert A (f32, [256,256]) into WMMA B-fragment order (f16).
// Fragment f = kc*16 + nt is the 32x16 tile  K in [kc*32, kc*32+32), N in
// [nt*16, nt*16+16).  B layout used here (16-bit, contiguous K):
//   lanes 0-15 : N = lane,      K = 2v + h        (v = vgpr 0..7, h = half)
//   lanes 16-31: N = lane - 16, K = 16 + 2v + h
// Stored per-lane contiguous: off = f*512 + lane*16 + (2v+h).
// ---------------------------------------------------------------------------
__global__ void isir_prep(const float* __restrict__ A,
                          _Float16* __restrict__ bfrag,
                          float* __restrict__ fracAcc) {
    int o    = blockIdx.x * 256 + threadIdx.x;      // 0 .. 65535
    int f    = o >> 9;
    int r    = o & 511;
    int lane = r >> 4;
    int q    = r & 15;                              // 2v + h
    int kc   = f >> 4;
    int nt   = f & 15;
    int Kl   = ((lane >> 4) << 4) + q;              // +16 for lanes 16..31
    int Nl   = lane & 15;
    bfrag[o] = (_Float16)A[(kc * 32 + Kl) * DDIM + nt * 16 + Nl];
    if (o == 0) *fracAcc = 0.f;
}

// ---------------------------------------------------------------------------
// Kernel 1: all proposal log-weights (up to a global additive constant).
//   w[tj, c] = -0.5*||(c1 + s*eps_row) @ A||^2 + 0.5*||eps_row||^2
// with c1 = q_loc - mu, s = exp(q_log_scale).
// One wave = 2 x 16 consecutive chains of one (t,j);
// 496 blocks * 8 waves * 4 tile-pairs = 31744 tiles.
// ---------------------------------------------------------------------------
__global__ void __launch_bounds__(256)
isir_gemm_w(const float* __restrict__ eps,
            const _Float16* __restrict__ bfrag,
            const float* __restrict__ mu,
            const float* __restrict__ q_loc,
            const float* __restrict__ qls,
            float* __restrict__ wAll) {
    extern __shared__ __align__(32) char smem[];
    _Float16* ldsB = (_Float16*)smem;                       // 65536 halfs (128 KB)
    float*    ldsC = (float*)(smem + 131072);               // c1[256]
    float*    ldsS = (float*)(smem + 131072 + 1024);        // scale[256]
    float*    wbuf = (float*)(smem + 131072 + 2048);        // 8 waves * 32 rows

    const int tid = threadIdx.x;

    // Stage B fragments (hot in L2) into LDS: 8192 uint4, 32 per thread.
    {
        const uint4* src = (const uint4*)bfrag;
        uint4*       dst = (uint4*)ldsB;
#pragma unroll
        for (int i = 0; i < 32; ++i) dst[tid + 256 * i] = src[tid + 256 * i];
    }
    ldsC[tid] = q_loc[tid] - mu[tid];
    ldsS[tid] = expf(qls[tid]);
    __syncthreads();

    const int wave = tid >> 5;
    const int lane = tid & 31;
    const int m    = lane & 15;     // row within 16-row tile
    const int grp  = lane >> 4;     // K-half select (contiguous K order)

    for (int p = 0; p < 4; ++p) {
        const int T0 = blockIdx.x * 64 + wave * 8 + p * 2; // even tile id
        const int tj = T0 >> 6;                            // (t*31 + j)
        const int c0 = (T0 & 63) << 4;                     // chain base (tile0)
        const float* rp0 = eps + ((size_t)tj * NC + (size_t)(c0 + m)) * DDIM;
        const float* rp1 = rp0 + 16 * DDIM;                // tile1: chains c0+16..c0+31

        v8f acc0[16], acc1[16];
        const v8f vz = {0.f, 0.f, 0.f, 0.f, 0.f, 0.f, 0.f, 0.f};
#pragma unroll
        for (int nt = 0; nt < 16; ++nt) { acc0[nt] = vz; acc1[nt] = vz; }
        float e2a = 0.f, e2b = 0.f;

        for (int kc = 0; kc < 8; ++kc) {
            const int kb = kc * 32 + grp * 16;
            // eps chunks, contiguous K order (matches bfrag):
            //   v0..7 -> K = kb + 0..15
            float4 a0 = *(const float4*)(rp0 + kb);
            float4 a1 = *(const float4*)(rp0 + kb + 4);
            float4 a2 = *(const float4*)(rp0 + kb + 8);
            float4 a3 = *(const float4*)(rp0 + kb + 12);
            float4 b0 = *(const float4*)(rp1 + kb);
            float4 b1 = *(const float4*)(rp1 + kb + 4);
            float4 b2 = *(const float4*)(rp1 + kb + 8);
            float4 b3 = *(const float4*)(rp1 + kb + 12);
            float4 c0v = *(const float4*)(ldsC + kb);
            float4 c1v = *(const float4*)(ldsC + kb + 4);
            float4 c2v = *(const float4*)(ldsC + kb + 8);
            float4 c3v = *(const float4*)(ldsC + kb + 12);
            float4 s0v = *(const float4*)(ldsS + kb);
            float4 s1v = *(const float4*)(ldsS + kb + 4);
            float4 s2v = *(const float4*)(ldsS + kb + 8);
            float4 s3v = *(const float4*)(ldsS + kb + 12);

            float eva[16] = {a0.x, a0.y, a0.z, a0.w, a1.x, a1.y, a1.z, a1.w,
                             a2.x, a2.y, a2.z, a2.w, a3.x, a3.y, a3.z, a3.w};
            float evb[16] = {b0.x, b0.y, b0.z, b0.w, b1.x, b1.y, b1.z, b1.w,
                             b2.x, b2.y, b2.z, b2.w, b3.x, b3.y, b3.z, b3.w};
            float cv[16] = {c0v.x, c0v.y, c0v.z, c0v.w, c1v.x, c1v.y, c1v.z, c1v.w,
                            c2v.x, c2v.y, c2v.z, c2v.w, c3v.x, c3v.y, c3v.z, c3v.w};
            float sv[16] = {s0v.x, s0v.y, s0v.z, s0v.w, s1v.x, s1v.y, s1v.z, s1v.w,
                            s2v.x, s2v.y, s2v.z, s2v.w, s3v.x, s3v.y, s3v.z, s3v.w};
            v16h va, vb;
#pragma unroll
            for (int e = 0; e < 16; ++e) {
                float xa = cv[e] + sv[e] * eva[e];
                float xb = cv[e] + sv[e] * evb[e];
                va[e] = (_Float16)xa;
                vb[e] = (_Float16)xb;
                e2a += eva[e] * eva[e];            // r == eps algebraically
                e2b += evb[e] * evb[e];
            }
#pragma unroll
            for (int nt = 0; nt < 16; ++nt) {
                const v16h b = *(const v16h*)(ldsB + (size_t)(kc * 16 + nt) * 512 + lane * 16);
                acc0[nt] = __builtin_amdgcn_wmma_f32_16x16x32_f16(
                    false, va, false, b, (short)0, acc0[nt], false, false);
                acc1[nt] = __builtin_amdgcn_wmma_f32_16x16x32_f16(
                    false, vb, false, b, (short)0, acc1[nt], false, false);
            }
        }

        // ||y||^2 per row: C/D layout -> element (r, lane) is M=r+8*grp, N=lane%16+16*nt.
        float s0[8], s1[8];
#pragma unroll
        for (int r = 0; r < 8; ++r) {
            float t0 = 0.f, t1 = 0.f;
#pragma unroll
            for (int nt = 0; nt < 16; ++nt) {
                float y0 = acc0[nt][r]; t0 += y0 * y0;
                float y1 = acc1[nt][r]; t1 += y1 * y1;
            }
            t0 += __shfl_xor(t0, 1, 32);
            t0 += __shfl_xor(t0, 2, 32);
            t0 += __shfl_xor(t0, 4, 32);
            t0 += __shfl_xor(t0, 8, 32);           // reduce within each 16-lane half
            t1 += __shfl_xor(t1, 1, 32);
            t1 += __shfl_xor(t1, 2, 32);
            t1 += __shfl_xor(t1, 4, 32);
            t1 += __shfl_xor(t1, 8, 32);
            s0[r] = t0;
            s1[r] = t1;
        }
        float e2ta = e2a + __shfl_xor(e2a, 16, 32);  // full ||eps||^2 for row m
        float e2tb = e2b + __shfl_xor(e2b, 16, 32);

        if ((lane & 15) == 0) {                    // lane 0 -> rows 0..7, lane 16 -> rows 8..15
            const int base = wave * 32 + grp * 8;
#pragma unroll
            for (int r = 0; r < 8; ++r) {
                wbuf[base + r]      = s0[r];
                wbuf[base + 16 + r] = s1[r];
            }
        }
        asm volatile("s_wait_dscnt 0x0" ::: "memory");  // wave-local DS is in-order
        float ysq0 = wbuf[wave * 32 + m];
        float ysq1 = wbuf[wave * 32 + 16 + m];
        if (lane < 16) {
            wAll[(size_t)tj * NC + c0 + lane]      = -0.5f * ysq0 + 0.5f * e2ta;
            wAll[(size_t)tj * NC + c0 + 16 + lane] = -0.5f * ysq1 + 0.5f * e2tb;
        }
    }
}

// ---------------------------------------------------------------------------
// Kernel 2: per-chain sequential gumbel-argmax scan (first-occurrence ties,
// strict >), recording source code per (t,c) and the n_eff fraction sum.
// ---------------------------------------------------------------------------
__global__ void isir_scan(const float* __restrict__ u,
                          const float* __restrict__ wAll,
                          int* __restrict__ srcArr,
                          float* __restrict__ fracAcc) {
    const int c = blockIdx.x * 256 + threadIdx.x;   // 1024 chains
    float wc = -INFINITY;
    int   srcCode = -1;                             // -1 => initial state (eps0)
    float cnt = 0.f;
    for (int t = 0; t < NSTEP; ++t) {
        float u0 = u[((size_t)t * 32 + 0) * NC + c];
        u0 = fminf(fmaxf(u0, 1e-12f), 1.f - 1e-12f);
        float best = wc + (-logf(-logf(u0)));
        int bi = 0;
        float bw = wc;
        for (int k = 1; k < 32; ++k) {
            float wv = wAll[((size_t)t * KM1 + (k - 1)) * NC + c];
            float uk = u[((size_t)t * 32 + k) * NC + c];
            uk = fminf(fmaxf(uk, 1e-12f), 1.f - 1e-12f);
            float v = wv + (-logf(-logf(uk)));
            if (v > best) { best = v; bi = k; bw = wv; }
        }
        if (bi != 0) { srcCode = t * KM1 + bi - 1; wc = bw; cnt += 1.f; }
        srcArr[t * NC + c] = srcCode;
    }
    __shared__ float red[256];
    red[threadIdx.x] = cnt;
    __syncthreads();
    for (int sft = 128; sft > 0; sft >>= 1) {
        if (threadIdx.x < sft) red[threadIdx.x] += red[threadIdx.x + sft];
        __syncthreads();
    }
    if (threadIdx.x == 0) atomicAdd(fracAcc, red[0] * (1.f / (float)NC));
}

// ---------------------------------------------------------------------------
// Kernel 3: materialize samples[t, c, :] = q_loc + exp(qls) * eps[src]
// (overwrites the wAll scratch region of d_out; scan already consumed it).
// ---------------------------------------------------------------------------
__global__ void isir_emit(const float* __restrict__ eps,
                          const float* __restrict__ eps0,
                          const float* __restrict__ q_loc,
                          const float* __restrict__ qls,
                          const int* __restrict__ srcArr,
                          float* __restrict__ out) {
    const int row = blockIdx.x;          // t*NC + c, 16384 rows
    const int c   = row & (NC - 1);
    const int d   = threadIdx.x;         // 256
    const int sc  = srcArr[row];
    const float* ep = (sc < 0) ? (eps0 + (size_t)c * DDIM)
                               : (eps + ((size_t)sc * NC + c) * DDIM);
    out[(size_t)row * DDIM + d] = q_loc[d] + expf(qls[d]) * ep[d];
}

__global__ void isir_final(const float* __restrict__ fracAcc,
                           float* __restrict__ out) {
    out[(size_t)NSTEP * NC * DDIM] = *fracAcc;
}

// ---------------------------------------------------------------------------
extern "C" void kernel_launch(void* const* d_in, const int* in_sizes, int n_in,
                              void* d_out, int out_size, void* d_ws, size_t ws_size,
                              hipStream_t stream) {
    const float* eps  = (const float*)d_in[0];   // [16,31,1024,256]
    const float* u    = (const float*)d_in[1];   // [16,32,1024]
    const float* eps0 = (const float*)d_in[2];   // [1024,256]
    const float* mu   = (const float*)d_in[3];   // [256]
    const float* A    = (const float*)d_in[4];   // [256,256]
    const float* qloc = (const float*)d_in[5];   // [256]
    const float* qls  = (const float*)d_in[6];   // [256]

    float* out = (float*)d_out;                  // samples [16,1024,256] + n_eff scalar

    _Float16* bfrag   = (_Float16*)d_ws;                          // 128 KB
    int*      srcArr  = (int*)((char*)d_ws + 131072);             // 64 KB
    float*    fracAcc = (float*)((char*)d_ws + 131072 + 65536);   // 4 B
    float*    wAll    = out;   // 2 MB scratch inside 16 MB d_out; overwritten by K3

    const size_t shmem = 131072 + 1024 + 1024 + 1024;  // B frags + c1 + scale + wbuf
    (void)hipFuncSetAttribute((const void*)isir_gemm_w,
                              hipFuncAttributeMaxDynamicSharedMemorySize,
                              (int)shmem);

    isir_prep<<<256, 256, 0, stream>>>(A, bfrag, fracAcc);
    isir_gemm_w<<<496, 256, shmem, stream>>>(eps, bfrag, mu, qloc, qls, wAll);
    isir_scan<<<4, 256, 0, stream>>>(u, wAll, srcArr, fracAcc);
    isir_emit<<<NSTEP * NC, DDIM, 0, stream>>>(eps, eps0, qloc, qls, srcArr, out);
    isir_final<<<1, 1, 0, stream>>>(fracAcc, out);
}
